// Masklayer_74981539053835
// MI455X (gfx1250) — 8-bit matrix operand order repaired and checked by reading, compiled, not run on hardware
//
#include <hip/hip_runtime.h>
#include <stdint.h>

typedef __attribute__((ext_vector_type(8))) int v8i;

#define N_ROWS    4096
#define D_COLS    64
#define ROW_BYTES 256
#define TILES     (N_ROWS / 16)

__global__ void canon_kernel(const float* __restrict__ x,
                             uint8_t* __restrict__ C,
                             uint32_t* __restrict__ norms,
                             uint32_t* __restrict__ dup) {
    int row = blockIdx.x * blockDim.x + threadIdx.x;
    if (row >= N_ROWS) return;
    const float* xr = x + (size_t)row * D_COLS;
    uint32_t* cr = (uint32_t*)(C + (size_t)row * ROW_BYTES);
    const uint32_t nanpat = 0x7FC00000u | (uint32_t)row;
    uint32_t nrm = 0;
    for (int k = 0; k < D_COLS; ++k) {
        float v = xr[k];
        uint32_t b;
        if (v != v)         b = nanpat;
        else if (v == 0.f)  b = 0u;
        else                b = __float_as_uint(v);
        cr[k] = b;
#pragma unroll
        for (int t = 0; t < 4; ++t) {
            uint32_t by = (b >> (8 * t)) & 0xFFu;
            nrm += by * by;
        }
    }
    norms[row] = nrm;
    dup[row]   = 0u;
}

__global__ void __launch_bounds__(32)
gram_dup_kernel(const uint8_t* __restrict__ C,
                const uint32_t* __restrict__ norms,
                uint32_t* __restrict__ dup) {
    const int I = blockIdx.x;
    const int J = blockIdx.y;
    if (J > I) return;

    const int lane = threadIdx.x;
    const int n    = lane & 15;
    const int half = lane >> 4;

    const uint8_t* arow = C + (size_t)(I * 16 + n) * ROW_BYTES;
    const uint8_t* brow = C + (size_t)(J * 16 + n) * ROW_BYTES;

    v8i acc = {0, 0, 0, 0, 0, 0, 0, 0};

#pragma unroll
    for (int kc = 0; kc < 4; ++kc) {
        const uint8_t* ap = arow + kc * 64 + half * 8;
        uint2 a01 = *(const uint2*)(ap);
        uint2 a23 = *(const uint2*)(ap + 16);
        uint2 a45 = *(const uint2*)(ap + 32);
        uint2 a67 = *(const uint2*)(ap + 48);
        v8i A;
        A[0] = (int)a01.x; A[1] = (int)a01.y;
        A[2] = (int)a23.x; A[3] = (int)a23.y;
        A[4] = (int)a45.x; A[5] = (int)a45.y;
        A[6] = (int)a67.x; A[7] = (int)a67.y;

        const uint8_t* bp = brow + kc * 64 + half * 8;
        uint2 b01 = *(const uint2*)(bp);
        uint2 b23 = *(const uint2*)(bp + 16);
        uint2 b45 = *(const uint2*)(bp + 32);
        uint2 b67 = *(const uint2*)(bp + 48);
        v8i B;
        B[0] = (int)b01.x; B[1] = (int)b01.y; B[2] = (int)b23.x; B[3] = (int)b23.y;
        B[4] = (int)b45.x; B[5] = (int)b45.y; B[6] = (int)b67.x; B[7] = (int)b67.y;

        acc = __builtin_amdgcn_wmma_i32_16x16x64_iu8(false, A, false, B, acc,
                                                     false, false);
    }

    const int j = J * 16 + n;
#pragma unroll
    for (int r = 0; r < 8; ++r) {
        const int i = I * 16 + half * 8 + r;
        const uint32_t ab = (uint32_t)acc[r];
        if (j < i && (2u * ab == norms[i] + norms[j])) {
            dup[i] = 1u;
        }
    }
}

__global__ void apply_kernel(const float* __restrict__ x,
                             const uint32_t* __restrict__ dup,
                             float* __restrict__ out) {
    int idx = blockIdx.x * blockDim.x + threadIdx.x;
    if (idx >= N_ROWS * D_COLS) return;
    int row = idx >> 6;
    out[idx] = dup[row] ? 0.0f : x[idx];
}

extern "C" void kernel_launch(void* const* d_in, const int* in_sizes, int n_in,
                              void* d_out, int out_size, void* d_ws, size_t ws_size,
                              hipStream_t stream) {
    (void)in_sizes; (void)n_in; (void)out_size; (void)ws_size;

    const float* x   = (const float*)d_in[0];
    float*       out = (float*)d_out;

    uint8_t*  C     = (uint8_t*)d_ws;
    uint32_t* norms = (uint32_t*)(C + (size_t)N_ROWS * ROW_BYTES);
    uint32_t* dup   = norms + N_ROWS;

    canon_kernel<<<dim3((N_ROWS + 255) / 256), dim3(256), 0, stream>>>(
        x, C, norms, dup);

    gram_dup_kernel<<<dim3(TILES, TILES), dim3(32), 0, stream>>>(
        C, norms, dup);

    apply_kernel<<<dim3((N_ROWS * D_COLS + 255) / 256), dim3(256), 0, stream>>>(
        x, dup, out);
}
